// SkipThoughtsNN_3616362463258
// MI455X (gfx1250) — compile-verified, matrix operand order checked by reading, not run on hardware
//
#include <hip/hip_runtime.h>

// ---------------------------------------------------------------------------
// Types / helpers
// ---------------------------------------------------------------------------
typedef __attribute__((ext_vector_type(16))) __bf16 v16bf;
typedef __attribute__((ext_vector_type(8)))  __bf16 v8bf;
typedef __attribute__((ext_vector_type(8)))  float  v8f;
typedef __attribute__((ext_vector_type(4)))  unsigned int v4u;
typedef __attribute__((ext_vector_type(8)))  unsigned int v8u;
typedef __attribute__((address_space(3)))    unsigned short lds_us;

#define B_SZ   64
#define T_SZ   128
#define E_SZ   620
#define E_PAD  640
#define HENC   600
#define HENC_P 608
#define NENC   1800
#define NENC_P 1824
#define HID    512
#define NDEC   1536
#define ROWS   (T_SZ * B_SZ)   // 8192

__device__ __forceinline__ unsigned short f2bf(float f) {
    unsigned int u = __float_as_uint(f);
    u += 0x7fffu + ((u >> 16) & 1u);           // round to nearest even
    return (unsigned short)(u >> 16);
}
__device__ __forceinline__ float bf2f(unsigned short h) {
    return __uint_as_float(((unsigned int)h) << 16);
}
__device__ __forceinline__ float sigmoidf(float x) {
    return 1.0f / (1.0f + __expf(-x));
}

union U16x8 { uint4 u4; v8bf v; };

// Load one 16x32 bf16 fragment (A-layout per ISA 7.12.2: lane<16 holds
// K{0..7,16..23}, lane>=16 holds K{8..15,24..31}; lane&15 = row).
// Works for global or LDS pointers (addrspace inferred after inlining).
__device__ __forceinline__ v16bf load_frag(const unsigned short* p, int ld,
                                           int r0, int k0, int lane) {
    int r  = lane & 15;
    int ko = (lane & 16) ? 8 : 0;
    const unsigned short* q = p + (size_t)(r0 + r) * ld + k0 + ko;
    U16x8 lo, hi;
    lo.u4 = *(const uint4*)(q);
    hi.u4 = *(const uint4*)(q + 16);
    return __builtin_shufflevector(lo.v, hi.v,
                                   0,1,2,3,4,5,6,7, 8,9,10,11,12,13,14,15);
}

__device__ __forceinline__ v8f wmma_bf16(v16bf a, v16bf b, v8f c) {
    return __builtin_amdgcn_wmma_f32_16x16x32_bf16(
        false, a, false, b, (short)0, c, false, false);
}

__device__ __forceinline__ v8f vzero8() {
    v8f z = {0.f,0.f,0.f,0.f,0.f,0.f,0.f,0.f};
    return z;
}

// Issue one TDM DMA of `elems` bf16 elements (contiguous) global -> LDS.
// D# layout per CDNA5 ISA (1-D tile, 2B elems).
// Must be executed by exactly one wave (TDM ignores EXEC).
__device__ __forceinline__ void tdm_load_1d(unsigned lds_base,
                                            const void* gptr,
                                            unsigned elems /* <= 65535 */) {
    unsigned long long ga = (unsigned long long)gptr;
    v4u g0;
    g0[0] = 1u;                                  // count=1, user descriptor
    g0[1] = lds_base;                            // lds_addr
    g0[2] = (unsigned)ga;                        // global_addr[31:0]
    g0[3] = ((unsigned)(ga >> 32) & 0x1FFFFFFu)  // global_addr[56:32]
          | (2u << 30);                          // type = image
    v8u g1;
    g1[0] = 1u << 16;                            // wg_mask=0, data_size=1 (2B)
    g1[1] = (elems & 0xFFFFu) << 16;             // tensor_dim0[15:0]
    g1[2] = ((elems >> 16) & 0xFFFFu)            // tensor_dim0[31:16]
          | (1u << 16);                          // tensor_dim1[15:0] = 1
    g1[3] = (elems & 0xFFFFu) << 16;             // tile_dim0 = elems
    g1[4] = 1u;                                  // tile_dim1 = 1, tile_dim2 = 0
    g1[5] = elems;                               // tensor_dim0_stride[31:0]
    g1[6] = 0u;
    g1[7] = 0u;
    asm volatile("tensor_load_to_lds %0, %1" :: "s"(g0), "s"(g1) : "memory");
    __builtin_amdgcn_s_wait_tensorcnt(0);
}

// ---------------------------------------------------------------------------
// Weight pack: fp32 [N][K] row-major -> bf16 [Npad][Kpad], zero padded
// ---------------------------------------------------------------------------
__global__ void k_pack_bf16(unsigned short* __restrict__ dst,
                            const float* __restrict__ src,
                            int N, int K, int Npad, int Kpad) {
    int idx = blockIdx.x * blockDim.x + threadIdx.x;
    int total = Npad * Kpad;
    if (idx >= total) return;
    int n = idx / Kpad, k = idx - n * Kpad;
    float v = (n < N && k < K) ? src[(size_t)n * K + k] : 0.f;
    dst[idx] = f2bf(v);
}

// ---------------------------------------------------------------------------
// Embedding gather -> X bf16 [T*B][E_PAD], t-major rows (row = t*64 + b)
// ---------------------------------------------------------------------------
__global__ void k_embed(unsigned short* __restrict__ X,
                        const int* __restrict__ tokens,
                        const float* __restrict__ we) {
    int idx = blockIdx.x * blockDim.x + threadIdx.x;
    if (idx >= ROWS * E_PAD) return;
    int row = idx / E_PAD, e = idx - row * E_PAD;
    int t = row >> 6, b = row & 63;
    int tok = tokens[b * T_SZ + t];
    float v = (e < E_SZ) ? we[(size_t)tok * E_SZ + e] : 0.f;
    X[idx] = f2bf(v);
}

__global__ void k_mask(int* __restrict__ mask_idx,
                       const int* __restrict__ tokens) {
    int b = threadIdx.x;
    if (b >= B_SZ) return;
    int mi = -1;
    for (int t = 0; t < T_SZ; ++t) {
        if (tokens[b * T_SZ + t] == 0) { mi = t - 1; break; }
    }
    mask_idx[b] = mi;
}

// ---------------------------------------------------------------------------
// Generic bf16 WMMA GEMM: C[M][ldc] = A[M][lda] * B^T (B packed [Npad][ldb])
// block = 256 threads (8 waves), wave tile 32x32, block tile 64x128.
// A block-tile (64 x Kpad, <= 80 KB) is DMA'd into LDS once via the TDM,
// then all WMMA A-fragments come from LDS; B streams from global/L2.
// ---------------------------------------------------------------------------
__global__ __launch_bounds__(256)
void k_gemm(const unsigned short* __restrict__ A,
            const unsigned short* __restrict__ Bw,
            int Kpad, int N, int lda, int ldb,
            const float* __restrict__ bias,
            unsigned short* __restrict__ Cbf,
            float* __restrict__ Cf, int ldc) {
    extern __shared__ unsigned short s_a[];      // [64][Kpad]
    int lane = threadIdx.x & 31;
    int wave = threadIdx.x >> 5;
    int mblk = blockIdx.y * 64;
    int m0 = (wave >> 2) * 32;                   // row offset inside LDS tile
    int n0 = blockIdx.x * 128 + (wave & 3) * 32;

    if (wave == 0) {
        unsigned lds_base = (unsigned)(unsigned long long)(lds_us*)s_a;
        tdm_load_1d(lds_base, A + (size_t)mblk * lda, (unsigned)(64 * Kpad));
    }
    __syncthreads();

    v8f acc00 = vzero8(), acc01 = vzero8(), acc10 = vzero8(), acc11 = vzero8();

    for (int k0 = 0; k0 < Kpad; k0 += 32) {
        v16bf a0 = load_frag(s_a, Kpad, m0,      k0, lane);
        v16bf a1 = load_frag(s_a, Kpad, m0 + 16, k0, lane);
        v16bf b0 = load_frag(Bw,  ldb,  n0,      k0, lane);
        v16bf b1 = load_frag(Bw,  ldb,  n0 + 16, k0, lane);
        acc00 = wmma_bf16(a0, b0, acc00);
        acc01 = wmma_bf16(a0, b1, acc01);
        acc10 = wmma_bf16(a1, b0, acc10);
        acc11 = wmma_bf16(a1, b1, acc11);
    }

    int mrow = (lane & 16) ? 8 : 0;
    int ncol = lane & 15;
    v8f accs[4] = {acc00, acc01, acc10, acc11};
    #pragma unroll
    for (int i = 0; i < 2; ++i) {
        #pragma unroll
        for (int j = 0; j < 2; ++j) {
            v8f a = accs[i * 2 + j];
            int nb = n0 + j * 16 + ncol;
            if (nb < N) {
                float bs = bias ? bias[nb] : 0.f;
                #pragma unroll
                for (int r = 0; r < 8; ++r) {
                    int mb = mblk + m0 + i * 16 + mrow + r;
                    float v = a[r] + bs;
                    size_t off = (size_t)mb * ldc + nb;
                    if (Cbf) Cbf[off] = f2bf(v);
                    else     Cf[off]  = v;
                }
            }
        }
    }
}

// ---------------------------------------------------------------------------
// Encoder GRU (both directions), one persistent workgroup per direction.
// h [64][608] and hp [64][1824] live in LDS (311 KB of the 320 KB WGP LDS).
// ---------------------------------------------------------------------------
__global__ __launch_bounds__(1024)
void k_enc_gru(const unsigned short* __restrict__ xp_f,
               const unsigned short* __restrict__ xp_b,
               const unsigned short* __restrict__ Whf,
               const unsigned short* __restrict__ Whb,
               const float* __restrict__ bhf,
               const float* __restrict__ bhb,
               unsigned short* __restrict__ hf_out,
               unsigned short* __restrict__ hb_out) {
    extern __shared__ unsigned short smem_e[];
    unsigned short* s_h  = smem_e;                        // [64][608]
    unsigned short* s_hp = smem_e + B_SZ * HENC_P;        // [64][1824]

    const int dir = blockIdx.x;
    const unsigned short* xp = dir ? xp_b : xp_f;
    const unsigned short* Wh = dir ? Whb  : Whf;
    const float* bh          = dir ? bhb  : bhf;
    unsigned short* hout     = dir ? hb_out : hf_out;

    const int tid = threadIdx.x, lane = tid & 31, wave = tid >> 5;

    for (int i = tid; i < B_SZ * HENC_P; i += 1024) s_h[i] = 0;
    __syncthreads();

    for (int t = 0; t < T_SZ; ++t) {
        int tt = dir ? (T_SZ - 1 - t) : t;

        // hp = h @ Wh^T   (M=64, N=1824, K=608)  456 tiles over 32 waves
        for (int tile = wave; tile < 4 * (NENC_P / 16); tile += 32) {
            int m0 = (tile & 3) * 16;
            int n0 = (tile >> 2) * 16;
            v8f acc = vzero8();
            for (int k0 = 0; k0 < HENC_P; k0 += 32) {
                v16bf af = load_frag(s_h, HENC_P, m0, k0, lane);
                v16bf bf = load_frag(Wh,  HENC_P, n0, k0, lane);
                acc = wmma_bf16(af, bf, acc);
            }
            int mrow = (lane & 16) ? 8 : 0, nc = lane & 15;
            #pragma unroll
            for (int r = 0; r < 8; ++r)
                s_hp[(size_t)(m0 + mrow + r) * NENC_P + n0 + nc] = f2bf(acc[r]);
        }
        __syncthreads();

        // gates (element-wise, fp32 math)
        const unsigned short* xrow = xp + (size_t)tt * B_SZ * NENC;
        for (int i = tid; i < B_SZ * HENC; i += 1024) {
            int bb = i / HENC, j = i - bb * HENC;
            const unsigned short* xr = xrow + (size_t)bb * NENC;
            float vxr = bf2f(xr[j]);
            float vxz = bf2f(xr[HENC + j]);
            float vxn = bf2f(xr[2 * HENC + j]);
            float hr = bf2f(s_hp[(size_t)bb * NENC_P + j])            + bh[j];
            float hz = bf2f(s_hp[(size_t)bb * NENC_P + HENC + j])     + bh[HENC + j];
            float hn = bf2f(s_hp[(size_t)bb * NENC_P + 2 * HENC + j]) + bh[2 * HENC + j];
            float r = sigmoidf(vxr + hr);
            float z = sigmoidf(vxz + hz);
            float n = tanhf(vxn + r * hn);
            float h = bf2f(s_h[bb * HENC_P + j]);
            float hnew = (1.f - z) * n + z * h;
            s_h[bb * HENC_P + j] = f2bf(hnew);
            hout[((size_t)tt * B_SZ + bb) * HENC + j] = f2bf(hnew);
        }
        __syncthreads();
    }
}

// ---------------------------------------------------------------------------
// Thought gather + conditional projection cp = thoughts @ C^T
// ---------------------------------------------------------------------------
__global__ void k_thoughts(float* __restrict__ th,
                           const unsigned short* __restrict__ hf,
                           const unsigned short* __restrict__ hb,
                           const int* __restrict__ mask_idx) {
    int idx = blockIdx.x * blockDim.x + threadIdx.x;
    if (idx >= B_SZ * 2 * HENC) return;
    int b = idx / (2 * HENC), j = idx - b * (2 * HENC);
    int mi = mask_idx[b];
    if (mi < 0) mi = 0;
    unsigned short v = (j < HENC)
        ? hf[((size_t)mi * B_SZ + b) * HENC + j]
        : hb[((size_t)mi * B_SZ + b) * HENC + (j - HENC)];
    th[idx] = bf2f(v);
}

__global__ void k_cp(float* __restrict__ cp,
                     const float* __restrict__ thoughts,
                     const float* __restrict__ C) {
    int idx = blockIdx.x * blockDim.x + threadIdx.x;
    if (idx >= B_SZ * NDEC) return;
    int b = idx / NDEC, j = idx - b * NDEC;
    const float* th = thoughts + (size_t)b * 1200;
    const float* c  = C + (size_t)j * 1200;
    float s = 0.f;
    for (int k = 0; k < 1200; ++k) s += th[k] * c[k];
    cp[idx] = s;
}

// ---------------------------------------------------------------------------
// Conditional GRU decoder (prev & next), one persistent workgroup each.
// Two-phase per step: (r,z) GEMM then (r*h)@U2 GEMM. 262 KB LDS.
// ---------------------------------------------------------------------------
__global__ __launch_bounds__(1024)
void k_dec_gru(const unsigned short* __restrict__ xdp,
               const unsigned short* __restrict__ xdn,
               const unsigned short* __restrict__ Up,
               const unsigned short* __restrict__ Un,
               const float* __restrict__ cpp,
               const float* __restrict__ cpn,
               unsigned short* __restrict__ hsp,
               unsigned short* __restrict__ hsn) {
    extern __shared__ unsigned short smem_d[];
    unsigned short* s_h  = smem_d;                     // [64][512]
    unsigned short* s_hp = smem_d + B_SZ * HID;        // [64][1024]
    unsigned short* s_rh = s_hp + B_SZ * 2 * HID;      // [64][512]

    const int dec = blockIdx.x;
    const unsigned short* xp = dec ? xdn : xdp;
    const unsigned short* U  = dec ? Un  : Up;
    const float* cp          = dec ? cpn : cpp;
    unsigned short* hs       = dec ? hsn : hsp;

    const int tid = threadIdx.x, lane = tid & 31, wave = tid >> 5;

    for (int i = tid; i < B_SZ * HID; i += 1024) s_h[i] = 0;
    __syncthreads();

    for (int t = 0; t < T_SZ; ++t) {
        const unsigned short* xrow = xp + (size_t)t * B_SZ * NDEC;

        // phase 1: hp = h @ [U0;U1]^T (N=1024, K=512), 256 tiles
        for (int tile = wave; tile < 4 * 64; tile += 32) {
            int m0 = (tile & 3) * 16;
            int n0 = (tile >> 2) * 16;
            v8f acc = vzero8();
            for (int k0 = 0; k0 < HID; k0 += 32) {
                v16bf af = load_frag(s_h, HID, m0, k0, lane);
                v16bf bf = load_frag(U,   HID, n0, k0, lane);
                acc = wmma_bf16(af, bf, acc);
            }
            int mrow = (lane & 16) ? 8 : 0, nc = lane & 15;
            #pragma unroll
            for (int r = 0; r < 8; ++r)
                s_hp[(size_t)(m0 + mrow + r) * (2 * HID) + n0 + nc] = f2bf(acc[r]);
        }
        __syncthreads();

        // gate 1: r, z ; stash z and r*h
        for (int i = tid; i < B_SZ * HID; i += 1024) {
            int bb = i >> 9, j = i & (HID - 1);
            float hp_r = bf2f(s_hp[(size_t)bb * 2 * HID + j]);
            float hp_z = bf2f(s_hp[(size_t)bb * 2 * HID + HID + j]);
            float r = sigmoidf(bf2f(xrow[(size_t)bb * NDEC + j]) + hp_r +
                               cp[(size_t)bb * NDEC + j]);
            float z = sigmoidf(bf2f(xrow[(size_t)bb * NDEC + HID + j]) + hp_z +
                               cp[(size_t)bb * NDEC + HID + j]);
            float h = bf2f(s_h[i]);
            s_rh[i] = f2bf(r * h);
            s_hp[(size_t)bb * 2 * HID + j] = f2bf(z);   // stash z
        }
        __syncthreads();

        // phase 2: hpn = (r*h) @ U2^T (N=512, K=512), 128 tiles
        for (int tile = wave; tile < 4 * 32; tile += 32) {
            int m0 = (tile & 3) * 16;
            int n0 = (tile >> 2) * 16;
            v8f acc = vzero8();
            for (int k0 = 0; k0 < HID; k0 += 32) {
                v16bf af = load_frag(s_rh, HID, m0, k0, lane);
                v16bf bf = load_frag(U + (size_t)2 * HID * HID, HID, n0, k0, lane);
                acc = wmma_bf16(af, bf, acc);
            }
            int mrow = (lane & 16) ? 8 : 0, nc = lane & 15;
            #pragma unroll
            for (int r = 0; r < 8; ++r)
                s_hp[(size_t)(m0 + mrow + r) * (2 * HID) + HID + n0 + nc] = f2bf(acc[r]);
        }
        __syncthreads();

        // gate 2: n, h' = (1-z)h + z n   (module's own update rule)
        for (int i = tid; i < B_SZ * HID; i += 1024) {
            int bb = i >> 9, j = i & (HID - 1);
            float n = tanhf(bf2f(xrow[(size_t)bb * NDEC + 2 * HID + j]) +
                            bf2f(s_hp[(size_t)bb * 2 * HID + HID + j]) +
                            cp[(size_t)bb * NDEC + 2 * HID + j]);
            float z = bf2f(s_hp[(size_t)bb * 2 * HID + j]);
            float h = bf2f(s_h[i]);
            float hnew = (1.f - z) * h + z * n;
            s_h[i] = f2bf(hnew);
            hs[((size_t)t * B_SZ + bb) * HID + j] = f2bf(hnew);
        }
        __syncthreads();
    }
}

// ---------------------------------------------------------------------------
// Host-side orchestration
// ---------------------------------------------------------------------------
extern "C" void kernel_launch(void* const* d_in, const int* in_sizes, int n_in,
                              void* d_out, int out_size, void* d_ws, size_t ws_size,
                              hipStream_t stream) {
    (void)in_sizes; (void)n_in; (void)out_size; (void)ws_size;

    const int*   tokens = (const int*)  d_in[0];
    const float* wembed = (const float*)d_in[1];
    const float* wx_f   = (const float*)d_in[2];
    const float* wh_f   = (const float*)d_in[3];
    const float* bx_f   = (const float*)d_in[4];
    const float* bh_f   = (const float*)d_in[5];
    const float* wx_b   = (const float*)d_in[6];
    const float* wh_b   = (const float*)d_in[7];
    const float* bx_b   = (const float*)d_in[8];
    const float* bh_b   = (const float*)d_in[9];
    const float* dp_W   = (const float*)d_in[10];
    const float* dp_U   = (const float*)d_in[11];
    const float* dp_C   = (const float*)d_in[12];
    const float* dn_W   = (const float*)d_in[13];
    const float* dn_U   = (const float*)d_in[14];
    const float* dn_C   = (const float*)d_in[15];
    const float* out_W  = (const float*)d_in[16];

    float* out_prev = (float*)d_out;
    float* out_next = out_prev + (size_t)ROWS * E_SZ;

    // bump allocator over d_ws
    char* ws = (char*)d_ws;
    size_t off = 0;
    auto alloc = [&](size_t bytes) -> void* {
        void* p = ws + off;
        off = (off + bytes + 255) & ~(size_t)255;
        return p;
    };
    typedef unsigned short us;
    us* Xbf  = (us*)alloc((size_t)ROWS * E_PAD * 2);
    us* Wxf  = (us*)alloc((size_t)1920 * E_PAD * 2);
    us* Wxb  = (us*)alloc((size_t)1920 * E_PAD * 2);
    us* Whf  = (us*)alloc((size_t)NENC_P * HENC_P * 2);
    us* Whb  = (us*)alloc((size_t)NENC_P * HENC_P * 2);
    us* Wdp  = (us*)alloc((size_t)NDEC * E_PAD * 2);
    us* Wdn  = (us*)alloc((size_t)NDEC * E_PAD * 2);
    us* Udp  = (us*)alloc((size_t)NDEC * HID * 2);
    us* Udn  = (us*)alloc((size_t)NDEC * HID * 2);
    us* Wout = (us*)alloc((size_t)640 * HID * 2);
    us* xpf  = (us*)alloc((size_t)ROWS * NENC * 2);
    us* xpb  = (us*)alloc((size_t)ROWS * NENC * 2);
    us* xdp  = (us*)alloc((size_t)ROWS * NDEC * 2);
    us* xdn  = (us*)alloc((size_t)ROWS * NDEC * 2);
    us* hf   = (us*)alloc((size_t)ROWS * HENC * 2);
    us* hb   = (us*)alloc((size_t)ROWS * HENC * 2);
    us* hsp  = (us*)alloc((size_t)ROWS * HID * 2);
    us* hsn  = (us*)alloc((size_t)ROWS * HID * 2);
    float* thoughts = (float*)alloc((size_t)B_SZ * 1200 * 4);
    float* cpp      = (float*)alloc((size_t)B_SZ * NDEC * 4);
    float* cpn      = (float*)alloc((size_t)B_SZ * NDEC * 4);
    int*   maskix   = (int*)alloc(B_SZ * 4);

    auto packN = [&](us* dst, const float* src, int N, int K, int Np, int Kp) {
        int total = Np * Kp;
        k_pack_bf16<<<(total + 255) / 256, 256, 0, stream>>>(dst, src, N, K, Np, Kp);
    };
    packN(Wxf, wx_f, NENC, E_SZ, 1920, E_PAD);
    packN(Wxb, wx_b, NENC, E_SZ, 1920, E_PAD);
    packN(Whf, wh_f, NENC, HENC, NENC_P, HENC_P);
    packN(Whb, wh_b, NENC, HENC, NENC_P, HENC_P);
    packN(Wdp, dp_W, NDEC, E_SZ, NDEC, E_PAD);
    packN(Wdn, dn_W, NDEC, E_SZ, NDEC, E_PAD);
    packN(Udp, dp_U, NDEC, HID, NDEC, HID);
    packN(Udn, dn_U, NDEC, HID, NDEC, HID);
    packN(Wout, out_W, E_SZ, HID, 640, HID);

    k_embed<<<(ROWS * E_PAD + 255) / 256, 256, 0, stream>>>(Xbf, tokens, wembed);
    k_mask<<<1, 64, 0, stream>>>(maskix, tokens);

    // Input projections (big WMMA GEMMs, TDM-staged A tiles)
    {
        dim3 g((NENC + 127) / 128, ROWS / 64);
        size_t lds = (size_t)64 * E_PAD * 2;
        k_gemm<<<g, 256, lds, stream>>>(Xbf, Wxf, E_PAD, NENC, E_PAD, E_PAD,
                                        bx_f, xpf, nullptr, NENC);
        k_gemm<<<g, 256, lds, stream>>>(Xbf, Wxb, E_PAD, NENC, E_PAD, E_PAD,
                                        bx_b, xpb, nullptr, NENC);
    }
    {
        dim3 g((NDEC + 127) / 128, ROWS / 64);
        size_t lds = (size_t)64 * E_PAD * 2;
        k_gemm<<<g, 256, lds, stream>>>(Xbf, Wdp, E_PAD, NDEC, E_PAD, E_PAD,
                                        nullptr, xdp, nullptr, NDEC);
        k_gemm<<<g, 256, lds, stream>>>(Xbf, Wdn, E_PAD, NDEC, E_PAD, E_PAD,
                                        nullptr, xdn, nullptr, NDEC);
    }

    // Encoder recurrence (both directions in parallel)
    size_t enc_lds = (size_t)(B_SZ * HENC_P + B_SZ * NENC_P) * 2;
    k_enc_gru<<<2, 1024, enc_lds, stream>>>(xpf, xpb, Whf, Whb, bh_f, bh_b, hf, hb);

    k_thoughts<<<(B_SZ * 2 * HENC + 255) / 256, 256, 0, stream>>>(thoughts, hf, hb, maskix);
    k_cp<<<(B_SZ * NDEC + 255) / 256, 256, 0, stream>>>(cpp, thoughts, dp_C);
    k_cp<<<(B_SZ * NDEC + 255) / 256, 256, 0, stream>>>(cpn, thoughts, dn_C);

    // Decoder recurrence (both decoders in parallel)
    size_t dec_lds = (size_t)(B_SZ * HID + B_SZ * 2 * HID + B_SZ * HID) * 2;
    k_dec_gru<<<2, 1024, dec_lds, stream>>>(xdp, xdn, Udp, Udn, cpp, cpn, hsp, hsn);

    // Output projections -> fp32 d_out (flat [T,B,E] == reference memory order)
    {
        dim3 g((E_SZ + 127) / 128, ROWS / 64);
        size_t lds = (size_t)64 * HID * 2;
        k_gemm<<<g, 256, lds, stream>>>(hsp, Wout, HID, E_SZ, HID, HID,
                                        nullptr, nullptr, out_prev, E_SZ);
        k_gemm<<<g, 256, lds, stream>>>(hsn, Wout, HID, E_SZ, HID, HID,
                                        nullptr, nullptr, out_next, E_SZ);
    }
}
